// AdasplashAttention_47167330845501
// MI455X (gfx1250) — matrix operand order repaired and checked by reading, not compiled, not run on hardware
//
#include <hip/hip_runtime.h>

// ---------------------------------------------------------------------------
// Causal entmax-1.5 attention for MI455X (gfx1250, wave32, WMMA + TDM).
//   B=4, H=16, S=1024, D=64, fp32 in/out.
//
// Per workgroup (4 waves, 64 queries, one (b,h)):
//   pass1: K chunk staged by the Tensor Data Mover (tensor_load_to_lds,
//          hardware-padded rows), S = mask(Q K^T / 16) via
//          v_wmma_f32_16x16x32_bf16, scores -> LDS (padded, 16B-aligned rows)
//   bisection: 20 iters for tau per row, 2 lanes/row, float4 LDS scans
//   pass2: P = relu(S - tau)^2 (bf16), O = (P V) / Z via WMMA
// ---------------------------------------------------------------------------

typedef __attribute__((ext_vector_type(16))) __bf16 v16bf;
typedef __attribute__((ext_vector_type(8)))  float  v8f;
typedef __attribute__((ext_vector_type(4)))  unsigned int u32x4;
typedef __attribute__((ext_vector_type(8)))  int    i32x8;
typedef __attribute__((ext_vector_type(4)))  int    i32x4;

#define NEGV (-1.0e30f)

constexpr int Bc = 4, Hc = 16, Sc = 1024, Dc = 64;
constexpr int SSTRIDE    = Sc + 4;       // score row stride (floats): 16B-aligned rows,
                                         // bank offset 4 per row -> conflict-free
constexpr int KS32_STRIDE = 68;          // TDM-staged K row stride (fp32 dwords)
constexpr int VT_STRIDE   = 40;          // staged V^T row stride (bf16) -> conflict-free
constexpr int WAVES = 4;
constexpr size_t SC_BYTES  = (size_t)WAVES * 16 * SSTRIDE * sizeof(float);   // 263168
constexpr size_t KV_BYTES  = 32 * KS32_STRIDE * sizeof(float);               // 8704 (>= V^T 5120)
constexpr size_t LDS_BYTES = SC_BYTES + KV_BYTES;                            // 271872 < 320KB

static __device__ __forceinline__ v8f wmma_bf(v16bf a, v16bf b, v8f c) {
  // D = A(16x32 bf16) * B(32x16 bf16) + C(16x16 f32)
  return __builtin_amdgcn_wmma_f32_16x16x32_bf16(false, a, false, b, (short)0, c,
                                                 false, false);
}

// Issue a TDM load of a 32-row x 64-dword fp32 tile (row-major, tensor row
// stride 64 dwords) into LDS at lds_off with padded LDS rows (stride 68 dwords).
static __device__ __forceinline__ void tdm_load_k_tile(const float* gsrc,
                                                       unsigned lds_off,
                                                       int rows_remaining) {
  const unsigned long long ga = (unsigned long long)(uintptr_t)gsrc;
  u32x4 g0;
  g0[0] = 1u;                                     // count=1, no gather
  g0[1] = lds_off;                                // D#.lds_addr
  g0[2] = (unsigned)ga;                           // global_addr[31:0]
  g0[3] = (unsigned)((ga >> 32) & 0x01FFFFFFu)    // global_addr[56:32]
        | (2u << 30);                             // type=2 ("image")
  i32x8 g1;
  g1[0] = (int)((2u << 16)                        // data_size = 4B
              | (1u << 20)                        // pad_enable
              | (5u << 22)                        // pad_interval: 64 dwords
              | (3u << 25));                      // pad_amount:   4 dwords
  g1[1] = (int)((Dc & 0xffff) << 16);             // tensor_dim0 = 64 (lo16)
  g1[2] = (int)(((unsigned)rows_remaining & 0xffff) << 16);  // dim0 hi=0, dim1 lo16
  g1[3] = (int)(((unsigned)Dc) << 16);            // dim1 hi=0, tile_dim0 = 64
  g1[4] = 32;                                     // tile_dim1 = 32, tile_dim2 = 0
  g1[5] = Dc;                                     // tensor_dim0_stride = 64 (lo32)
  g1[6] = 0;                                      // stride0 hi, stride1 lo
  g1[7] = 0;                                      // stride1 hi
  const i32x4 gz4 = {0, 0, 0, 0};
  const i32x8 gz8 = {0, 0, 0, 0, 0, 0, 0, 0};
  // 6-arg toolchain variant: (g0, g1, g2, g3, extra, cpol)
  __builtin_amdgcn_tensor_load_to_lds(g0, g1, gz4, gz4, gz8, 0);
  __builtin_amdgcn_s_wait_tensorcnt(0);
}

__global__ __launch_bounds__(128)
void entmax15_attn_kernel(const float* __restrict__ Q, const float* __restrict__ K,
                          const float* __restrict__ V, float* __restrict__ O) {
  extern __shared__ char smem[];
  float*  sc   = (float*)smem;                    // [4 waves][16 rows][SSTRIDE]
  float*  kf32 = (float*)(smem + SC_BYTES);       // pass1: TDM-staged K chunk (fp32)
  __bf16* vstg = (__bf16*)(smem + SC_BYTES);      // pass2: V^T chunk (bf16)

  const int tid  = threadIdx.x;
  const int wave = tid >> 5;
  const int lane = tid & 31;
  const int r    = lane & 15;                     // row / N index within 16-group
  const int g    = lane >> 4;                     // lane half
  const int qblk = blockIdx.x;                    // 64-query block
  const int bh   = blockIdx.y;                    // flat (b*H + h)
  const int q0   = qblk * 64 + wave * 16;         // this wave's first query

  const float* Qh = Q + (size_t)bh * Sc * Dc;
  const float* Kh = K + (size_t)bh * Sc * Dc;
  const float* Vh = V + (size_t)bh * Sc * Dc;
  float*       Oh = O + (size_t)bh * Sc * Dc;

  // ---- A fragments: Q rows, scaled by 0.5*sm_scale = 1/16, bf16 ------------
  // A layout (16-bit 16x32): lane (r,g), element e -> K = (e&7) + (e>>3)*16 + g*8
  v16bf aq0, aq1;
#pragma unroll
  for (int e = 0; e < 16; ++e) {
    const int d = (e & 7) + ((e >> 3) << 4) + g * 8;
    aq0[e] = (__bf16)(Qh[(size_t)(q0 + r) * Dc + d]      * 0.0625f);
    aq1[e] = (__bf16)(Qh[(size_t)(q0 + r) * Dc + 32 + d] * 0.0625f);
  }

  const int nkt_wg = qblk * 2 + 2;            // 32-key chunks needed by workgroup
  const int nkt_wv = (q0 + 47) >> 5;          // chunks needed by this wave (causal)
  float* scw = sc + (size_t)wave * 16 * SSTRIDE;

  // ===================== pass 1: scores -> LDS ==============================
  for (int kt = 0; kt < nkt_wg; ++kt) {
    const int kb = kt << 5;
    __syncthreads();                          // previous chunk fully consumed
    if (wave == 0) {                          // wave-uniform: TDM issue + wait
      tdm_load_k_tile(Kh + (size_t)kb * Dc,
                      (unsigned)(uintptr_t)kf32,   // low 32 bits = LDS offset
                      Sc - kb);
    }
    __syncthreads();                          // tile visible to all waves

    if (kt < nkt_wv) {
#pragma unroll
      for (int sub = 0; sub < 2; ++sub) {     // two 16-key sub-tiles
        // B layout (32x16): lane (n=r,g), element e -> Kdim = (e&7) + (e>>3)*16 + g*8
        const float* kr = kf32 + (sub * 16 + r) * KS32_STRIDE + g * 8;
        v16bf b0, b1;
#pragma unroll
        for (int e = 0; e < 16; ++e) {
          b0[e] = (__bf16)kr[(e & 7) + ((e >> 3) << 4)];
          b1[e] = (__bf16)kr[32 + (e & 7) + ((e >> 3) << 4)];
        }
        v8f c = {0.f, 0.f, 0.f, 0.f, 0.f, 0.f, 0.f, 0.f};
        c = wmma_bf(aq0, b0, c);              // D-dims 0..31
        c = wmma_bf(aq1, b1, c);              // D-dims 32..63
        // C layout: VGPR v -> row m = v + g*8, col n = r
#pragma unroll
        for (int v = 0; v < 8; ++v) {
          const int m   = v + g * 8;
          const int col = kb + sub * 16 + r;
          scw[m * SSTRIDE + col] = (col <= q0 + m) ? c[v] : NEGV;
        }
      }
    }
  }

  // ===================== bisection for tau (wave-local) =====================
  // Scan the padded region [0, kpad): masked cols hold NEG (contribute 0).
  // kpad is wave-uniform -> fully converged loops around the shuffles.
  const int kpad = nkt_wv << 5;
  const int half = kpad >> 1;                      // multiple of 16
  const float* srow = scw + r * SSTRIDE;           // lanes r and r+16 share row r
  const int cbeg = g ? half : 0;
  const int cend = g ? kpad : half;

  float mx = NEGV;
  for (int c = cbeg; c < cend; c += 4) {
    const float4 x = *(const float4*)(srow + c);
    mx = fmaxf(mx, fmaxf(fmaxf(x.x, x.y), fmaxf(x.z, x.w)));
  }
  mx = fmaxf(mx, __shfl_xor(mx, 16, 32));

  float lo = mx - 1.0f, hi = mx;
#pragma unroll 1
  for (int it = 0; it < 20; ++it) {
    const float t = 0.5f * (lo + hi);
    float acc = 0.0f;
    for (int c = cbeg; c < cend; c += 4) {
      const float4 x = *(const float4*)(srow + c);
      const float d0 = fmaxf(x.x - t, 0.0f);
      const float d1 = fmaxf(x.y - t, 0.0f);
      const float d2 = fmaxf(x.z - t, 0.0f);
      const float d3 = fmaxf(x.w - t, 0.0f);
      acc = fmaf(d0, d0, acc);
      acc = fmaf(d1, d1, acc);
      acc = fmaf(d2, d2, acc);
      acc = fmaf(d3, d3, acc);
    }
    acc += __shfl_xor(acc, 16, 32);
    if (acc >= 1.0f) lo = t; else hi = t;          // f(t) = acc - 1
  }
  const float tau = 0.5f * (lo + hi);

  // ===================== pass 2: O = (P V) / Z ==============================
  v8f acc[4];
#pragma unroll
  for (int n0 = 0; n0 < 4; ++n0)
    acc[n0] = (v8f){0.f, 0.f, 0.f, 0.f, 0.f, 0.f, 0.f, 0.f};
  float zloc = 0.0f;

  for (int kt = 0; kt < nkt_wg; ++kt) {
    const int kb = kt << 5;
    __syncthreads();
    // stage 32x64 V chunk TRANSPOSED (Vt[d][key]) as bf16, padded
    for (int i = tid; i < 1024; i += 128) {
      const int key = i >> 5, dp = (i & 31) << 1;
      const float2 f2 = ((const float2*)(Vh + (size_t)(kb + key) * Dc))[dp >> 1];
      vstg[dp * VT_STRIDE + key]       = (__bf16)f2.x;
      vstg[(dp + 1) * VT_STRIDE + key] = (__bf16)f2.y;
    }
    if (kt + 1 < nkt_wg)   // pull next V chunk toward L2 (global_prefetch_b8)
      __builtin_prefetch(Vh + (size_t)(kb + 32) * Dc + tid * 16, 0, 1);
    __syncthreads();

    if (kt < nkt_wv) {
      // P fragment (A layout, row = r): p = relu(s - tau)^2, bf16-rounded
      const float* pr = srow + kb + g * 8;         // 16B aligned
      const float4 s0 = *(const float4*)(pr);
      const float4 s1 = *(const float4*)(pr + 4);
      const float4 s2 = *(const float4*)(pr + 16);
      const float4 s3 = *(const float4*)(pr + 20);
      const float sv[16] = {s0.x, s0.y, s0.z, s0.w, s1.x, s1.y, s1.z, s1.w,
                            s2.x, s2.y, s2.z, s2.w, s3.x, s3.y, s3.z, s3.w};
      v16bf pa;
#pragma unroll
      for (int e = 0; e < 16; ++e) {
        const float d  = fmaxf(sv[e] - tau, 0.0f);
        const __bf16 pb = (__bf16)(d * d);
        zloc += (float)pb;                         // Z from the rounded weights
        pa[e] = pb;
      }
#pragma unroll
      for (int n0 = 0; n0 < 4; ++n0) {
        // B = V chunk (32 keys x 16 dims): element e -> key = (e&7) + (e>>3)*16 + g*8, dim = n0*16 + r
        const __bf16* vr = vstg + (n0 * 16 + r) * VT_STRIDE + g * 8;
        v16bf bv;
#pragma unroll
        for (int e = 0; e < 16; ++e) bv[e] = vr[(e & 7) + ((e >> 3) << 4)];
        acc[n0] = wmma_bf(pa, bv, acc[n0]);
      }
    }
  }

  // exact renormalization and store
  const float zc = zloc + __shfl_xor(zloc, 16, 32);   // lane l holds Z[row l&15]
#pragma unroll
  for (int v = 0; v < 8; ++v) {
    const int   m = v + g * 8;
    const float w = 1.0f / __shfl(zc, m, 32);
#pragma unroll
    for (int n0 = 0; n0 < 4; ++n0)
      Oh[(size_t)(q0 + m) * Dc + n0 * 16 + r] = acc[n0][v] * w;
  }
}

// ---------------------------------------------------------------------------
extern "C" void kernel_launch(void* const* d_in, const int* in_sizes, int n_in,
                              void* d_out, int out_size, void* d_ws, size_t ws_size,
                              hipStream_t stream) {
  (void)in_sizes; (void)n_in; (void)out_size; (void)d_ws; (void)ws_size;
  const float* q = (const float*)d_in[0];
  const float* k = (const float*)d_in[1];
  const float* v = (const float*)d_in[2];
  float* o = (float*)d_out;

  dim3 grid(Sc / 64, Bc * Hc);   // 16 query-blocks x 64 heads = 1024 workgroups
  dim3 block(128);               // 4 waves (wave32)
  hipLaunchKernelGGL(entmax15_attn_kernel, grid, block, LDS_BYTES, stream,
                     q, k, v, o);
}
